// AttentionDecoupleMetric_77146202570971
// MI455X (gfx1250) — matrix operand order repaired and checked by reading, not compiled, not run on hardware
//
#include <hip/hip_runtime.h>
#include <hip/hip_bf16.h>

typedef __attribute__((ext_vector_type(16))) _Float16 v16h;
typedef __attribute__((ext_vector_type(8)))  _Float16 v8h;
typedef __attribute__((ext_vector_type(8)))  float    v8f;

#define NB 16
#define NC 512
#define NP 784
#define QP 800   /* NP padded to multiple of 32 (WMMA K chunk) */

// ---------------------------------------------------------------------------
// Stage 1: pairwise L1 distance D[b,p,q] = sum_c |x[b,c,p] - x[b,c,q]|
// x layout is [B, C, P] (P contiguous). Tile: 64 p-rows x 32 q-cols, 8-channel
// chunks staged in LDS for the p-rows; q column values loaded coalesced
// directly (adjacent lanes -> adjacent p addresses). D stored as f16 with the
// q dimension zero-padded to QP so the WMMA stage needs no remainder logic.
// ---------------------------------------------------------------------------
__global__ void pairdist_kernel(const float* __restrict__ x,
                                _Float16* __restrict__ D) {
    const int tid = threadIdx.x;          // 256 threads
    const int lx  = tid & 31;             // q lane within tile
    const int ly  = tid >> 5;             // p group 0..7
    const int q0  = blockIdx.x * 32;      // 0..768 (25 tiles -> q up to 800)
    const int p0  = blockIdx.y * 64;      // 0..768 (13 tiles, last partial)
    const int b   = blockIdx.z;

    const int q  = q0 + lx;
    const int qc = (q < NP) ? q : (NP - 1);   // clamp; padded cols stored as 0

    __shared__ float sP[8][64];
    const float* xb = x + (size_t)b * NC * NP;

    float acc[8];
#pragma unroll
    for (int i = 0; i < 8; ++i) acc[i] = 0.f;

    for (int cc = 0; cc < NC; cc += 8) {
        __syncthreads();
#pragma unroll
        for (int r = 0; r < 2; ++r) {           // 512 floats, 2 per thread
            int idx = tid + r * 256;
            int c   = idx >> 6;
            int pp  = idx & 63;
            int gp  = p0 + pp; if (gp >= NP) gp = NP - 1;  // clamped rows unused
            sP[c][pp] = xb[(size_t)(cc + c) * NP + gp];
        }
        __syncthreads();

        float qv[8];
#pragma unroll
        for (int c = 0; c < 8; ++c)
            qv[c] = xb[(size_t)(cc + c) * NP + qc];
#pragma unroll
        for (int c = 0; c < 8; ++c) {
#pragma unroll
            for (int i = 0; i < 8; ++i)
                acc[i] += fabsf(sP[c][ly * 8 + i] - qv[c]);
        }
    }

#pragma unroll
    for (int i = 0; i < 8; ++i) {
        int gp = p0 + ly * 8 + i;
        if (gp < NP)
            D[(size_t)(b * NP + gp) * QP + q] =
                (q < NP) ? (_Float16)acc[i] : (_Float16)0.f;
    }
}

// ---------------------------------------------------------------------------
// Stage 2: per-row inverse L1 norm. One wave per row (8 waves / block).
// ---------------------------------------------------------------------------
__global__ void rowsum_kernel(const _Float16* __restrict__ D,
                              float* __restrict__ invS) {
    const int lane = threadIdx.x & 31;
    const int wave = threadIdx.x >> 5;
    const int row  = blockIdx.x * 8 + wave;      // 0 .. 16*784-1
    const _Float16* r = D + (size_t)row * QP;
    float s = 0.f;
    for (int k = lane; k < QP; k += 32) s += (float)r[k];
#pragma unroll
    for (int o = 16; o > 0; o >>= 1) s += __shfl_down(s, o, 32);
    if (lane == 0) invS[row] = 1.0f / fmaxf(s, 1e-12f);
}

// ---------------------------------------------------------------------------
// Stage 3 init: v0 = ones/P (zero in the padded tail); zero the ping buffer's
// tail too (it is never written by the mat-vec, but WMMA B reads it).
// ---------------------------------------------------------------------------
__global__ void initv_kernel(_Float16* __restrict__ vA,
                             _Float16* __restrict__ vB) {
    int t = blockIdx.x * 256 + threadIdx.x;
    if (t < NB * QP) {
        int q = t % QP;
        vA[t] = (q < NP) ? (_Float16)(1.0f / (float)NP) : (_Float16)0.f;
        vB[t] = (_Float16)0.f;
    }
}

// ---------------------------------------------------------------------------
// Stage 3: v' = diag(invS) * D * v  via v_wmma_f32_16x16x32_f16.
// One wave per (batch, 16-row tile). A = 16x32 f16 tile of D (documented A
// layout: lane<16 -> K 0..7 & 16..23, lane>=16 -> K 8..15 & 24..31).
// B = 32x16 with every column equal to v (same K layout as A: lane<16 holds
// K=0..7 & 16..23, lane>=16 K 8..15 & 24..31; N = lane&15 is irrelevant since columns are replicated).
// After 25 chunks, lanes 0 / 16 hold column N=0 -> rows p0..p0+7 / p0+8..+15.
// ---------------------------------------------------------------------------
__global__ void matvec_wmma_kernel(const _Float16* __restrict__ D,
                                   const float* __restrict__ invS,
                                   const _Float16* __restrict__ vin,
                                   _Float16* __restrict__ vout,
                                   float* __restrict__ out,
                                   int write_out) {
    const int pt   = blockIdx.x;            // 0..48
    const int b    = blockIdx.y;            // 0..15
    const int lane = threadIdx.x;           // 0..31 (one full wave, EXEC all-1)
    const int p0   = pt * 16;
    const int m    = lane & 15;
    const bool lo16 = (lane < 16);

    const _Float16* Drow = D + (size_t)(b * NP + p0 + m) * QP;
    const _Float16* vb   = vin + b * QP;

    v8f c = {};
    for (int j = 0; j < 25; ++j) {
        const int q0    = j * 32;
        const int abase = q0 + (lo16 ? 0 : 8);
        v8h alo = *(const v8h*)(Drow + abase);        // K +0..7 (of this half)
        v8h ahi = *(const v8h*)(Drow + abase + 16);   // K +16..23
        const int bbase = q0 + (lo16 ? 0 : 8);
        v8h blo = *(const v8h*)(vb + bbase);
        v8h bhi = *(const v8h*)(vb + bbase + 16);
        v16h a, bm;
#pragma unroll
        for (int i = 0; i < 8; ++i) {
            a[i]      = alo[i];
            a[i + 8]  = ahi[i];
            bm[i]     = blo[i];
            bm[i + 8] = bhi[i];
        }
        // (neg_a, A, neg_b, B, c_mod, C, reuse_a, reuse_b)
        c = __builtin_amdgcn_wmma_f32_16x16x32_f16(
                false, a, false, bm, (short)0, c, false, false);
    }

    if ((lane & 15) == 0) {                 // N == 0 column holders
        const int rbase = p0 + (lo16 ? 0 : 8);
#pragma unroll
        for (int r = 0; r < 8; ++r) {
            int p = rbase + r;
            float val = c[r] * invS[b * NP + p];
            vout[b * QP + p] = (_Float16)val;
            if (write_out) out[b * NP + p] = val;
        }
    }
}

// ---------------------------------------------------------------------------
// Workspace layout (bytes):
//   D    f16 : 16*784*800*2 = 20,070,400
//   invS f32 : 16*784*4     =     50,176   @ 20,070,400
//   vA   f16 : 16*800*2     =     25,600   @ 20,120,576
//   vB   f16 : 16*800*2     =     25,600   @ 20,146,176
// total ~20.2 MB (D stays resident in the 192 MB L2 across the 10 mat-vecs).
// ---------------------------------------------------------------------------
extern "C" void kernel_launch(void* const* d_in, const int* in_sizes, int n_in,
                              void* d_out, int out_size, void* d_ws, size_t ws_size,
                              hipStream_t stream) {
    const float* x = (const float*)d_in[0];
    float* out = (float*)d_out;

    char* ws = (char*)d_ws;
    _Float16* D    = (_Float16*)ws;
    float*    invS = (float*)(ws + 20070400);
    _Float16* vA   = (_Float16*)(ws + 20120576);
    _Float16* vB   = (_Float16*)(ws + 20146176);

    // Stage 1: pairwise L1 distances (dominant cost, VALU-bound, LDS-tiled)
    pairdist_kernel<<<dim3(QP / 32, (NP + 63) / 64, NB), 256, 0, stream>>>(x, D);

    // Stage 2: row sums -> inverse norms
    rowsum_kernel<<<dim3((NB * NP) / 8), 256, 0, stream>>>(D, invS);

    // Stage 3: v0 = u, then ten WMMA mat-vecs; last one writes d_out
    initv_kernel<<<dim3((NB * QP + 255) / 256), 256, 0, stream>>>(vA, vB);
    for (int k = 0; k < 10; ++k) {
        const _Float16* src = (k & 1) ? vB : vA;
        _Float16*       dst = (k & 1) ? vA : vB;
        matvec_wmma_kernel<<<dim3(NP / 16, NB), 32, 0, stream>>>(
            D, invS, src, dst, out, (k == 9) ? 1 : 0);
    }
}
